// Seq2SeqSumm_90099823935883
// MI455X (gfx1250) — matrix operand order repaired and checked by reading, not compiled, not run on hardware
//
#include <hip/hip_runtime.h>
#include <hip/hip_bf16.h>

// ---------------------------------------------------------------------------
// Seq2Seq summarizer forward for MI455X (gfx1250, wave32, WMMA).
// All GEMMs use v_wmma_f32_16x16x32_bf16 (bf16 in / f32 accumulate).
// Encoder: 2 persistent workgroups (fwd/bwd LSTM), [x|h] staged in LDS.
// Decoder: 1 persistent workgroup, 64 sequential attention steps.
// Round 2: waves own NT n-tiles sharing one A fragment per K-block, so each
// load-wait covers NT WMMAs and LDS A-traffic drops NT x (latency-critical
// for the sequential recurrence).
// ---------------------------------------------------------------------------

typedef __bf16 bf16;
typedef __bf16 v16bf __attribute__((ext_vector_type(16)));
typedef float  v8f   __attribute__((ext_vector_type(8)));

#define TSRC  1024
#define BATCH 32
#define HID   512
#define EMB   512      // encoder input dim (2*E)
#define G4    2048     // 4*H gate width
#define KCAT  1024     // concat [x|h] K dim
#define T_TGT 64
#define OUTD  256      // H/2
#define SPECIAL 4

__device__ __forceinline__ v8f zero8() {
  v8f z;
#pragma unroll
  for (int i = 0; i < 8; ++i) z[i] = 0.f;
  return z;
}

#define WMMA_BF16(a, b, c) \
  __builtin_amdgcn_wmma_f32_16x16x32_bf16(false, (a), false, (b), (short)0, (c), false, false)

// A fragment (16x32 bf16, M x K) from a row-major source. rowp = &A[row][0].
// Layout: lanes 0-15 row M=lane hold K=0..15, lanes 16-31 hold K=16..31
// (contiguous, matching load_b_colmajor).
__device__ __forceinline__ v16bf load_a_rowptr(const bf16* rowp, int k0) {
  const int lane = threadIdx.x & 31;
  const int kb = (lane >> 4) << 4;       // 0 or 16
  const bf16* p = rowp + k0 + kb;
  v16bf a;
#pragma unroll
  for (int v = 0; v < 4; ++v) { a[2 * v] = p[2 * v]; a[2 * v + 1] = p[2 * v + 1]; }
#pragma unroll
  for (int v = 0; v < 4; ++v) { a[8 + 2 * v] = p[8 + 2 * v]; a[9 + 2 * v] = p[9 + 2 * v]; }
  return a;
}

// B fragment (32x16 bf16, K x N) from W stored [N][K] ("col-major B" == torch [out,in]).
// lanes 0-15 hold K=0..15, lanes 16-31 hold K=16..31; column N = n + lane%16.
// Contiguous 32B per lane -> one wide load.
__device__ __forceinline__ v16bf load_b_colmajor(const bf16* W, int K, int n, int k0) {
  const int lane = threadIdx.x & 31;
  const int col = n + (lane & 15);
  const int kb = (lane >> 4) << 4;       // 0 or 16
  const bf16* p = W + (size_t)col * K + k0 + kb;
  return *(const v16bf*)p;
}

// C/D: VGPR v holds row m + v + 8*(lane/16), col n + lane%16
__device__ __forceinline__ void store_c(float* C, int ldc, int m, int n, v8f acc) {
  const int lane = threadIdx.x & 31;
  const int col = n + (lane & 15);
  const int r0 = m + ((lane >> 4) << 3);
#pragma unroll
  for (int v = 0; v < 8; ++v) C[(size_t)(r0 + v) * ldc + col] = acc[v];
}

// Wave-cooperative GEMM: C[M,N] = A[M,K](row-major, LDS or global) x W[N,K].
// Each wave owns groups of NT adjacent 16x16 n-tiles sharing one A fragment
// per K-block: per kb -> 1 LDS A load + NT clause-grouped B loads + NT WMMAs.
// Uniform control flow per wave (EXEC all ones for WMMA).
template <int NT>
__device__ __forceinline__ void wave_gemm(const bf16* A, int lda, const bf16* W, int K,
                                          int M, int N, float* C, int ldc, int nwaves) {
  const int wave = threadIdx.x >> 5;
  const int lane = threadIdx.x & 31;
  const int ngrp = N / (16 * NT);
  const int groups = (M >> 4) * ngrp;
  for (int g = wave; g < groups; g += nwaves) {
    const int m = (g / ngrp) << 4;
    const int n0 = (g % ngrp) * (16 * NT);
    v8f acc[NT];
#pragma unroll
    for (int j = 0; j < NT; ++j) acc[j] = zero8();
    const bf16* arow = A + (size_t)(m + (lane & 15)) * lda;
    for (int kb = 0; kb < K; kb += 32) {
      const v16bf a = load_a_rowptr(arow, kb);
      v16bf bf[NT];
#pragma unroll
      for (int j = 0; j < NT; ++j) bf[j] = load_b_colmajor(W, K, n0 + 16 * j, kb);
#pragma unroll
      for (int j = 0; j < NT; ++j) acc[j] = WMMA_BF16(a, bf[j], acc[j]);
    }
#pragma unroll
    for (int j = 0; j < NT; ++j) store_c(C, ldc, m, n0 + 16 * j, acc[j]);
  }
}

__device__ __forceinline__ float sigf(float x) { return 1.f / (1.f + __expf(-x)); }

// ------------------------------ pack kernels -------------------------------

__global__ void k_cvt_bf16(const float* __restrict__ in, bf16* __restrict__ out, int n) {
  int i = blockIdx.x * blockDim.x + threadIdx.x;
  if (i < n) out[i] = (bf16)in[i];
}

// out[N][Kx+Kh] = [Wx | Wh] row-concat (torch weights are [out,in], already B-col-major)
__global__ void k_pack_cat(const float* __restrict__ Wx, const float* __restrict__ Wh,
                           bf16* __restrict__ out, int Kx, int Kh, int N) {
  const int K = Kx + Kh;
  int i = blockIdx.x * blockDim.x + threadIdx.x;
  if (i >= N * K) return;
  int nn = i / K, k = i % K;
  out[i] = (bf16)(k < Kx ? Wx[(size_t)nn * Kx + k] : Wh[(size_t)nn * Kh + (k - Kx)]);
}

// in: f32 [K][N]  ->  out: bf16 [N][K]  (transpose to B-col-major)
__global__ void k_transpose_bf(const float* __restrict__ in, bf16* __restrict__ out,
                               int K, int N) {
  int i = blockIdx.x * blockDim.x + threadIdx.x;
  if (i >= K * N) return;
  int nn = i / K, k = i % K;
  out[i] = (bf16)in[(size_t)k * N + nn];
}

// --------------------------- encoder (persistent) --------------------------

__global__ __launch_bounds__(1024) void k_encoder(
    const bf16* __restrict__ art,           // [TSRC][B][EMB] bf16
    const bf16* __restrict__ Wcat,          // [2][G4][KCAT]
    const float* __restrict__ b_f, const float* __restrict__ b_b,
    const float* __restrict__ ih, const float* __restrict__ ic,   // [2][H]
    bf16* __restrict__ outbf_f, bf16* __restrict__ outbf_b,       // [TSRC][B][H]
    float* __restrict__ cstate,             // [2][B][H]
    float* __restrict__ gatesbuf)           // [2][B][G4]
{
  __shared__ bf16 xh[BATCH][KCAT];          // [x_t | h] staging, 64 KB
  const int dir = blockIdx.x;
  const int tid = threadIdx.x;
  const bf16* W = Wcat + (size_t)dir * G4 * KCAT;
  const float* bias = dir ? b_b : b_f;
  bf16* outp = dir ? outbf_b : outbf_f;
  float* cst = cstate + (size_t)dir * BATCH * HID;
  float* gates = gatesbuf + (size_t)dir * BATCH * G4;

  for (int i = tid; i < BATCH * HID; i += 1024) {
    int h = i & 511;
    xh[i >> 9][EMB + h] = (bf16)ih[dir * HID + h];
    cst[i] = ic[dir * HID + h];
  }
  __threadfence();
  __syncthreads();

  for (int t = 0; t < TSRC; ++t) {
    const int ts = dir ? (TSRC - 1 - t) : t;          // backward scans reversed
    const bf16* x = art + (size_t)ts * BATCH * EMB;
    for (int i = tid; i < BATCH * EMB; i += 1024) xh[i >> 9][i & 511] = x[i];
    __syncthreads();

    // gates = [x|h] @ [Wih|Whh]^T  (K=1024, N=2048): 64 groups of 4 tiles
    wave_gemm<4>(&xh[0][0], KCAT, W, KCAT, BATCH, G4, gates, G4, 32);
    __threadfence();
    __syncthreads();

    bf16* orow = outp + (size_t)t * BATCH * HID;      // scan-order outputs
    for (int i = tid; i < BATCH * HID; i += 1024) {
      int b = i >> 9, h = i & 511;
      const float* g = gates + (size_t)b * G4;
      float gi = g[h] + bias[h];
      float gf = g[512 + h] + bias[512 + h];
      float gg = g[1024 + h] + bias[1024 + h];
      float go = g[1536 + h] + bias[1536 + h];
      float c = sigf(gf) * cst[i] + sigf(gi) * tanhf(gg);
      float hn = sigf(go) * tanhf(c);
      cst[i] = c;
      orow[i] = (bf16)hn;
      xh[b][EMB + h] = (bf16)hn;                      // feed next step
    }
    __threadfence();
    __syncthreads();
  }
}

// final_h = concat(hT_fwd, hT_bwd), final_c = concat(cT) -> bf16 [B][2H]
__global__ void k_finalcat(const bf16* __restrict__ hf, const bf16* __restrict__ hb,
                           const float* __restrict__ c, bf16* __restrict__ hcat,
                           bf16* __restrict__ ccat) {
  int i = blockIdx.x * blockDim.x + threadIdx.x;     // [B][H]
  int b = i >> 9, h = i & 511;
  hcat[(size_t)b * 1024 + h] = hf[i];
  hcat[(size_t)b * 1024 + 512 + h] = hb[i];
  ccat[(size_t)b * 1024 + h] = (bf16)c[i];
  ccat[(size_t)b * 1024 + 512 + h] = (bf16)c[BATCH * HID + i];
}

// one 16x16 tile per wave, grid sized exactly
__global__ __launch_bounds__(256) void k_gemm_tile(
    const bf16* __restrict__ A, int lda, const bf16* __restrict__ W, int K, int N,
    float* __restrict__ C, int ldc) {
  const int gid = (blockIdx.x * 256 + threadIdx.x) >> 5;
  const int lane = threadIdx.x & 31;
  const int nt = N >> 4;
  const int m = (gid / nt) << 4;
  const int n = (gid % nt) << 4;
  v8f acc = zero8();
  const bf16* arow = A + (size_t)(m + (lane & 15)) * lda;
  for (int kb = 0; kb < K; kb += 32)
    acc = WMMA_BF16(load_a_rowptr(arow, kb), load_b_colmajor(W, K, n, kb), acc);
  store_c(C, ldc, m, n, acc);
}

// attention[b,t,h] = enc_art[t,b,:] . wm[:,h]   (A gathered from fwd/bwd outputs)
// 8 waves/block; each wave owns 4 adjacent n-tiles sharing one A fragment/kb.
__global__ __launch_bounds__(256) void k_attention(
    const bf16* __restrict__ of, const bf16* __restrict__ ob,
    const bf16* __restrict__ wmT,            // [H][2H]
    bf16* __restrict__ att)                  // [B][TSRC][H]
{
  const int mtile = blockIdx.x;              // 2048 m-tiles over T*B rows
  const int wave = threadIdx.x >> 5, lane = threadIdx.x & 31;
  const int t = mtile >> 1;
  const int b0 = (mtile & 1) << 4;
  const int rb = b0 + (lane & 15);
  const bf16* rowf = of + ((size_t)t * BATCH + rb) * HID;
  const bf16* rowb = ob + ((size_t)(TSRC - 1 - t) * BATCH + rb) * HID;  // un-reverse
  const int n0 = wave << 6;                  // 4 tiles x 16 cols per wave
  v8f acc[4];
#pragma unroll
  for (int j = 0; j < 4; ++j) acc[j] = zero8();
  for (int kb = 0; kb < 512; kb += 32) {
    const v16bf a = load_a_rowptr(rowf, kb);
#pragma unroll
    for (int j = 0; j < 4; ++j)
      acc[j] = WMMA_BF16(a, load_b_colmajor(wmT, 1024, n0 + 16 * j, kb), acc[j]);
  }
  for (int kb = 512; kb < 1024; kb += 32) {
    const v16bf a = load_a_rowptr(rowb, kb - 512);
#pragma unroll
    for (int j = 0; j < 4; ++j)
      acc[j] = WMMA_BF16(a, load_b_colmajor(wmT, 1024, n0 + 16 * j, kb), acc[j]);
  }
  const int r0 = b0 + ((lane >> 4) << 3);
#pragma unroll
  for (int j = 0; j < 4; ++j) {
    const int col = n0 + 16 * j + (lane & 15);
#pragma unroll
    for (int v = 0; v < 8; ++v)
      att[((size_t)(r0 + v) * TSRC + t) * HID + col] = (bf16)acc[j][v];
  }
}

__global__ void k_seqmean(const bf16* __restrict__ att, const int* __restrict__ lens,
                          float* __restrict__ sm) {
  const int b = blockIdx.x;
  const int len = lens[b];
  for (int h = threadIdx.x; h < HID; h += blockDim.x) {
    float s = 0.f;
    const bf16* p = att + (size_t)b * TSRC * HID + h;
    for (int t = 0; t < len; ++t) s += (float)p[(size_t)t * HID];
    sm[(size_t)b * HID + h] = s / (float)len;
  }
}

// --------------------------- decoder (persistent) --------------------------

__global__ __launch_bounds__(1024) void k_decoder(
    const float* __restrict__ emb, const int* __restrict__ art_lens,
    const bf16* __restrict__ Wcat,           // [G4][KCAT]
    const float* __restrict__ dec_b,
    const bf16* __restrict__ wqT,            // [H][H]
    const bf16* __restrict__ W1T,            // [H][2H]
    const float* __restrict__ b1,
    const bf16* __restrict__ W2T,            // [OUTD][H]
    const bf16* __restrict__ att,            // [B][TSRC][H]
    const float* __restrict__ seqmean,
    float* __restrict__ hstate, float* __restrict__ cstate,   // init_h/init_c in-place
    float* __restrict__ gates, float* __restrict__ query,
    float* __restrict__ scores, float* __restrict__ probs,
    float* __restrict__ tmp1, float* __restrict__ prev_out,
    float* __restrict__ dec_all, float* __restrict__ h_all, float* __restrict__ c_all)
{
  __shared__ bf16 xh[BATCH][KCAT];
  const int tid = threadIdx.x;

  // ---- prologue: init_out = project([init_h | seq_mean]) ----
  for (int i = tid; i < BATCH * HID; i += 1024) {
    int b = i >> 9, h = i & 511;
    xh[b][h] = (bf16)hstate[i];
    xh[b][512 + h] = (bf16)seqmean[i];
  }
  __syncthreads();
  wave_gemm<2>(&xh[0][0], KCAT, W1T, 1024, BATCH, HID, tmp1, HID, 32);
  __threadfence();
  __syncthreads();
  for (int i = tid; i < BATCH * HID; i += 1024)
    xh[i >> 9][i & 511] = (bf16)tanhf(tmp1[i] + b1[i & 511]);
  __syncthreads();
  wave_gemm<1>(&xh[0][0], KCAT, W2T, HID, BATCH, OUTD, prev_out, OUTD, 32);
  __threadfence();
  __syncthreads();

  for (int step = 0; step < T_TGT; ++step) {
    const float* embrow = emb + (size_t)(SPECIAL + step) * OUTD;
    // stage LSTM input [emb_i | prev_out | h]
    for (int i = tid; i < BATCH * OUTD; i += 1024) {
      int b = i >> 8, j = i & 255;
      xh[b][j] = (bf16)embrow[j];
      xh[b][OUTD + j] = (bf16)prev_out[i];
    }
    for (int i = tid; i < BATCH * HID; i += 1024)
      xh[i >> 9][512 + (i & 511)] = (bf16)hstate[i];
    __syncthreads();

    wave_gemm<4>(&xh[0][0], KCAT, Wcat, KCAT, BATCH, G4, gates, G4, 32);
    __threadfence();
    __syncthreads();

    for (int i = tid; i < BATCH * HID; i += 1024) {    // LSTM cell
      int b = i >> 9, h = i & 511;
      const float* g = gates + (size_t)b * G4;
      float gi = g[h] + dec_b[h];
      float gf = g[512 + h] + dec_b[512 + h];
      float gg = g[1024 + h] + dec_b[1024 + h];
      float go = g[1536 + h] + dec_b[1536 + h];
      float c = sigf(gf) * cstate[i] + sigf(gi) * tanhf(gg);
      float hn = sigf(go) * tanhf(c);
      cstate[i] = c;
      hstate[i] = hn;
      h_all[((size_t)b * T_TGT + step) * HID + h] = hn;
      c_all[((size_t)b * T_TGT + step) * HID + h] = c;
      xh[b][h] = (bf16)hn;                             // A for query+project
    }
    __threadfence();
    __syncthreads();

    wave_gemm<2>(&xh[0][0], KCAT, wqT, HID, BATCH, HID, query, HID, 32);  // query = h@Wq
    __threadfence();
    __syncthreads();

    {  // scores: thread = t, loop b  (batched matvec, VALU)
      const int t = tid;
      for (int b = 0; b < BATCH; ++b) {
        const bf16* arow = att + ((size_t)b * TSRC + t) * HID;
        const float* q = query + (size_t)b * HID;
        float s = 0.f;
        for (int k = 0; k < HID; ++k) s += q[k] * (float)arow[k];
        scores[(size_t)b * TSRC + t] = (t < art_lens[b]) ? s : -1e9f;
      }
    }
    __threadfence();
    __syncthreads();

    {  // softmax: wave w handles batch row b=w via shuffle reductions
      const int w = tid >> 5, lane = tid & 31;
      float sl[32];
      float mx = -1e30f;
#pragma unroll
      for (int j = 0; j < 32; ++j) {
        sl[j] = scores[(size_t)w * TSRC + lane + 32 * j];
        mx = fmaxf(mx, sl[j]);
      }
      for (int m = 16; m; m >>= 1) mx = fmaxf(mx, __shfl_xor(mx, m, 32));
      float sum = 0.f;
#pragma unroll
      for (int j = 0; j < 32; ++j) { sl[j] = __expf(sl[j] - mx); sum += sl[j]; }
      for (int m = 16; m; m >>= 1) sum += __shfl_xor(sum, m, 32);
      const float inv = 1.f / sum;
#pragma unroll
      for (int j = 0; j < 32; ++j) probs[(size_t)w * TSRC + lane + 32 * j] = sl[j] * inv;
    }
    __threadfence();
    __syncthreads();

    for (int i = tid; i < BATCH * HID; i += 1024) {    // context -> xh[:,512:]
      int b = i >> 9, h = i & 511;
      const float* p = probs + (size_t)b * TSRC;
      const bf16* ab = att + (size_t)b * TSRC * HID + h;
      float s = 0.f;
      for (int t = 0; t < TSRC; ++t) s += p[t] * (float)ab[(size_t)t * HID];
      xh[b][512 + h] = (bf16)s;
    }
    __syncthreads();

    wave_gemm<2>(&xh[0][0], KCAT, W1T, 1024, BATCH, HID, tmp1, HID, 32);  // project
    __threadfence();
    __syncthreads();
    for (int i = tid; i < BATCH * HID; i += 1024)
      xh[i >> 9][i & 511] = (bf16)tanhf(tmp1[i] + b1[i & 511]);
    __syncthreads();
    wave_gemm<1>(&xh[0][0], KCAT, W2T, HID, BATCH, OUTD, prev_out, OUTD, 32);
    __threadfence();
    __syncthreads();
    for (int i = tid; i < BATCH * OUTD; i += 1024) {
      int b = i >> 8, o = i & 255;
      dec_all[((size_t)b * T_TGT + step) * OUTD + o] = prev_out[i];
    }
    __syncthreads();
  }
}

// ------------------------------- launcher ----------------------------------

extern "C" void kernel_launch(void* const* d_in, const int* in_sizes, int n_in,
                              void* d_out, int out_size, void* d_ws, size_t ws_size,
                              hipStream_t stream) {
  const float* article = (const float*)d_in[0];    // [1024,32,512]
  const int* art_lens = (const int*)d_in[1];       // [32]
  const float* emb = (const float*)d_in[3];        // [V,256]
  const float* eWihF = (const float*)d_in[4], *eWhhF = (const float*)d_in[5];
  const float* ebF = (const float*)d_in[6];
  const float* eWihB = (const float*)d_in[7], *eWhhB = (const float*)d_in[8];
  const float* ebB = (const float*)d_in[9];
  const float* initH = (const float*)d_in[10], *initC = (const float*)d_in[11];
  const float* dec_hW = (const float*)d_in[12], *dec_cW = (const float*)d_in[13];
  const float* attn_wm = (const float*)d_in[14], *attn_wq = (const float*)d_in[15];
  const float* pW1 = (const float*)d_in[16], *pb1 = (const float*)d_in[17];
  const float* pW2 = (const float*)d_in[18];
  const float* dWih = (const float*)d_in[19], *dWhh = (const float*)d_in[20];
  const float* db = (const float*)d_in[21];

  // workspace carve-out (~150 MB total)
  size_t off = 0;
  auto take = [&](size_t bytes) -> char* {
    char* p = (char*)d_ws + off;
    off += (bytes + 255) & ~(size_t)255;
    return p;
  };
  bf16* art_bf   = (bf16*)take((size_t)TSRC * BATCH * EMB * 2);
  bf16* WcatEnc  = (bf16*)take((size_t)2 * G4 * KCAT * 2);
  bf16* WcatDec  = (bf16*)take((size_t)G4 * KCAT * 2);
  bf16* wmT      = (bf16*)take((size_t)HID * 1024 * 2);
  bf16* wqT      = (bf16*)take((size_t)HID * HID * 2);
  bf16* W1T      = (bf16*)take((size_t)HID * 1024 * 2);
  bf16* W2T      = (bf16*)take((size_t)OUTD * HID * 2);
  bf16* hWT      = (bf16*)take((size_t)HID * 1024 * 2);
  bf16* cWT      = (bf16*)take((size_t)HID * 1024 * 2);
  bf16* outbf_f  = (bf16*)take((size_t)TSRC * BATCH * HID * 2);
  bf16* outbf_b  = (bf16*)take((size_t)TSRC * BATCH * HID * 2);
  float* cstEnc  = (float*)take((size_t)2 * BATCH * HID * 4);
  float* gatEnc  = (float*)take((size_t)2 * BATCH * G4 * 4);
  bf16* hcat     = (bf16*)take((size_t)BATCH * 1024 * 2);
  bf16* ccat     = (bf16*)take((size_t)BATCH * 1024 * 2);
  bf16* att_bf   = (bf16*)take((size_t)BATCH * TSRC * HID * 2);
  float* seqmean = (float*)take((size_t)BATCH * HID * 4);
  float* hstate  = (float*)take((size_t)BATCH * HID * 4);
  float* cstDec  = (float*)take((size_t)BATCH * HID * 4);
  float* gatDec  = (float*)take((size_t)BATCH * G4 * 4);
  float* query   = (float*)take((size_t)BATCH * HID * 4);
  float* scores  = (float*)take((size_t)BATCH * TSRC * 4);
  float* probs   = (float*)take((size_t)BATCH * TSRC * 4);
  float* tmp1    = (float*)take((size_t)BATCH * HID * 4);
  float* prevout = (float*)take((size_t)BATCH * OUTD * 4);

  float* dec_all = (float*)d_out;                              // [B,64,256]
  float* h_all = dec_all + (size_t)BATCH * T_TGT * OUTD;       // [B,64,512]
  float* c_all = h_all + (size_t)BATCH * T_TGT * HID;          // [B,64,512]

  // phase 0: weight packing / precision conversion
  {
    int n = TSRC * BATCH * EMB;
    k_cvt_bf16<<<n / 256, 256, 0, stream>>>(article, art_bf, n);
  }
  k_pack_cat<<<(G4 * KCAT) / 256, 256, 0, stream>>>(eWihF, eWhhF, WcatEnc, EMB, HID, G4);
  k_pack_cat<<<(G4 * KCAT) / 256, 256, 0, stream>>>(eWihB, eWhhB,
                                                    WcatEnc + (size_t)G4 * KCAT, EMB, HID, G4);
  k_pack_cat<<<(G4 * KCAT) / 256, 256, 0, stream>>>(dWih, dWhh, WcatDec, HID, HID, G4);
  k_transpose_bf<<<(1024 * 512) / 256, 256, 0, stream>>>(attn_wm, wmT, 1024, 512);
  k_transpose_bf<<<(512 * 512) / 256, 256, 0, stream>>>(attn_wq, wqT, 512, 512);
  k_transpose_bf<<<(1024 * 512) / 256, 256, 0, stream>>>(pW1, W1T, 1024, 512);
  k_transpose_bf<<<(512 * 256) / 256, 256, 0, stream>>>(pW2, W2T, 512, 256);
  k_transpose_bf<<<(1024 * 512) / 256, 256, 0, stream>>>(dec_hW, hWT, 1024, 512);
  k_transpose_bf<<<(1024 * 512) / 256, 256, 0, stream>>>(dec_cW, cWT, 1024, 512);

  // phase 1: bidirectional encoder (persistent, one block per direction)
  k_encoder<<<2, 1024, 0, stream>>>(art_bf, WcatEnc, ebF, ebB, initH, initC,
                                    outbf_f, outbf_b, cstEnc, gatEnc);

  // phase 2: decoder initial states + attention memory + seq mean
  k_finalcat<<<(BATCH * HID) / 256, 256, 0, stream>>>(
      outbf_f + (size_t)(TSRC - 1) * BATCH * HID,
      outbf_b + (size_t)(TSRC - 1) * BATCH * HID, cstEnc, hcat, ccat);
  k_gemm_tile<<<8, 256, 0, stream>>>(hcat, 1024, hWT, 1024, 512, hstate, 512);
  k_gemm_tile<<<8, 256, 0, stream>>>(ccat, 1024, cWT, 1024, 512, cstDec, 512);
  k_attention<<<(TSRC * BATCH) / 16, 256, 0, stream>>>(outbf_f, outbf_b, wmT, att_bf);
  k_seqmean<<<BATCH, 256, 0, stream>>>(att_bf, art_lens, seqmean);

  // phase 3: sequential attention decoder (persistent single block)
  k_decoder<<<1, 1024, 0, stream>>>(emb, art_lens, WcatDec, db, wqT, W1T, pb1, W2T,
                                    att_bf, seqmean, hstate, cstDec, gatDec, query,
                                    scores, probs, tmp1, prevout, dec_all, h_all, c_all);

  (void)in_sizes; (void)n_in; (void)out_size; (void)ws_size;
}
